// DMDNet_2430951490164
// MI455X (gfx1250) — matrix operand order repaired and checked by reading, compiled, not run on hardware
//
#include <hip/hip_runtime.h>
#include <hip/hip_bf16.h>
#include <stdint.h>

// ---------------------------------------------------------------------------
// CDNA5 (gfx1250) implicit-GEMM 3x3 conv on v_wmma_f32_16x16x32_bf16.
// Activations live in packed-bf16 "PK" layout: dword[pair p][pixel] holds
// channels (2p, 2p+1) as {lo,hi} bf16 -- exactly the WMMA A-operand dword.
// Producers zero-fill channel padding up to the 32-wide K-chunk, so the conv
// hot loop is: 8 dword loads (immediate offsets via TW specialization) +
// 8 cndmask + 2 ds_load_b128 + 1 v_wmma per K-chunk, unrolled x2.
// fp32 copies are emitted only for residual sources and concat outputs.
// ---------------------------------------------------------------------------

typedef __attribute__((ext_vector_type(16))) __bf16 v16bf;
typedef __attribute__((ext_vector_type(8)))  float  v8f;

union BFVec { v16bf v; unsigned u[8]; };

__device__ __forceinline__ unsigned pk2bf(float a, float b) {
  // round-to-nearest-even fp32 -> bf16, packed {hi=b, lo=a}
  unsigned ua = __float_as_uint(a);
  unsigned ub = __float_as_uint(b);
  ua = (ua + 0x7FFFu + ((ua >> 16) & 1u)) >> 16;
  ub = (ub + 0x7FFFu + ((ub >> 16) & 1u)) >> 16;
  return (ub << 16) | (ua & 0xFFFFu);
}

// TW > 0: compile-time square image width (offsets become immediates).
// TW == 0: generic runtime Hin/Win path (small Q-patch convs).
template<int TW>
__global__ __launch_bounds__(128)
void conv3x3_pk_wmma(const unsigned* __restrict__ Xpk, const float* __restrict__ Wt,
                     const float* __restrict__ Bias, const float* __restrict__ Res,
                     float* __restrict__ Yf, unsigned* __restrict__ Ypk,
                     int CinR, int CinP, int Cout,
                     int Hin, int Win, int Hout, int Wout,
                     int stride, int dil, int leaky,
                     long xB, long yfB, long ypkB, long resB)
{
  // packed bf16 weight pairs; word((t,)g,n,j) = (g*16+n)*8 + j,
  // g = kc*2 + half, K-pair p = g*8 + j
  __shared__ unsigned wlds[12288];   // 48 KB

  const int WinC = (TW > 0) ? TW : Win;
  const int HinC = (TW > 0) ? TW : Hin;
  const int HWin = WinC * HinC;

  const int tid  = threadIdx.x;
  const int lane = tid & 31;
  const int wave = tid >> 5;
  const int nL   = lane & 15;       // N column (cout) for B/C/D
  const int half = lane >> 4;       // K-half (A/B), M-half (C/D)

  const int b       = blockIdx.z;
  const int nBase   = blockIdx.y * 16;
  const int HW      = Hout * Wout;
  const int pixTile = blockIdx.x * 64 + wave * 16;

  const int  kchunks = CinP >> 5;                // CinP is a multiple of 32
  const int  nldsw   = kchunks * 256;            // LDS words per tap
  const bool ldsAll  = (nldsw * 9) <= 12288;     // all 9 taps resident?

  const unsigned* Xb = Xpk + (long)b * xB;

  // A-matrix row (pixel) for this lane: M = lane%16
  const int  pixA = pixTile + nL;
  const bool pAok = pixA < HW;
  const int  oy   = pAok ? (pixA / Wout) : 0;
  const int  ox   = pAok ? (pixA - oy * Wout) : 0;

  v8f acc;
  {
    const float bv = Bias[nBase + nL];
    #pragma unroll
    for (int r = 0; r < 8; ++r) acc[r] = bv;
  }

  // ---- weight staging: fp32 OIHW -> packed bf16 pairs in LDS -------------
  auto stageTap = [&](int t, int base) {
    for (int idx = tid; idx < nldsw; idx += 128) {
      const int j = idx & 7;
      const int n = (idx >> 3) & 15;
      const int g = idx >> 7;
      const int p  = g * 8 + j;                 // cin pair index
      const int c0 = p * 2, c1 = c0 + 1;
      const long wb = ((long)(nBase + n) * CinR) * 9 + t;   // OIHW 3x3
      const float w0 = (c0 < CinR) ? Wt[wb + (long)c0 * 9] : 0.f;
      const float w1 = (c1 < CinR) ? Wt[wb + (long)c1 * 9] : 0.f;
      wlds[base + idx] = pk2bf(w0, w1);
    }
  };

  if (ldsAll) {
    for (int t = 0; t < 9; ++t) stageTap(t, t * nldsw);
    __syncthreads();
  }

  for (int t = 0; t < 9; ++t) {
    if (!ldsAll) {
      __syncthreads();
      stageTap(t, 0);
      __syncthreads();
    }
    const unsigned* wtap = wlds + (ldsAll ? t * nldsw : 0);

    const int kh = t / 3, kw = t - kh * 3;
    const int  iy    = oy * stride + (kh - 1) * dil;   // pad == dil
    const int  ix    = ox * stride + (kw - 1) * dil;
    const bool inImg = pAok & (iy >= 0) & (iy < HinC) & (ix >= 0) & (ix < WinC);
    // clamped per-lane base: always a valid address, masked after load
    const unsigned* Xp = Xb + (inImg ? ((long)iy * WinC + ix) : 0l);
    __builtin_prefetch(Xp, 0, 1);

    auto doChunk = [&](int kc) {
      BFVec a, w;
      #pragma unroll
      for (int j = 0; j < 8; ++j) {
        // A operand in this code's contiguous k order: VGPR j holds K pair; lanes16-31 shift K by 16
        const int pb = j + 8 * half;
        const unsigned raw = Xp[(kc * 16 + pb) * HWin];   // imm offsets if TW>0
        a.u[j] = inImg ? raw : 0u;
      }
      const uint4* wv =
          (const uint4*)(wtap + ((((kc * 2 + half) << 4) + nL) << 3));
      *(uint4*)&w.u[0] = wv[0];
      *(uint4*)&w.u[4] = wv[1];
      acc = __builtin_amdgcn_wmma_f32_16x16x32_bf16(false, a.v, false, w.v,
                                                    (short)0, acc, false, false);
    };

    int kc = 0;
    for (; kc + 2 <= kchunks; kc += 2) {   // x2 unroll (kchunks even on hot layers)
      doChunk(kc);
      doChunk(kc + 1);
    }
    if (kc < kchunks) doChunk(kc);
  }

  // ---- epilogue: LeakyReLU / residual / fp32 store / PK-bf16 store --------
  const int  cO      = nBase + nL;
  const bool aligned = ((HW & 3) == 0) && (pixTile + 16 <= HW);
  const long pixOff  = pixTile + 8 * half;     // reg r -> pixel pixOff + r

  float v[8];
  #pragma unroll
  for (int r = 0; r < 8; ++r) {
    float tv = acc[r];
    if (leaky) tv = tv > 0.f ? tv : 0.2f * tv;
    v[r] = tv;
  }

  if (Res) {
    const float* Rp = Res + (long)b * resB + (long)cO * HW + pixOff;
    if (aligned) {
      const float4 r0 = *(const float4*)Rp;
      const float4 r1 = *(const float4*)(Rp + 4);
      v[0] += r0.x; v[1] += r0.y; v[2] += r0.z; v[3] += r0.w;
      v[4] += r1.x; v[5] += r1.y; v[6] += r1.z; v[7] += r1.w;
    } else {
      #pragma unroll
      for (int r = 0; r < 8; ++r)
        if (pixOff + r < HW) v[r] += Rp[r];
    }
  }

  if (Yf) {
    float* Yp = Yf + (long)b * yfB + (long)cO * HW + pixOff;
    if (aligned) {
      *(float4*)Yp       = make_float4(v[0], v[1], v[2], v[3]);
      *(float4*)(Yp + 4) = make_float4(v[4], v[5], v[6], v[7]);
    } else {
      #pragma unroll
      for (int r = 0; r < 8; ++r)
        if (pixOff + r < HW) Yp[r] = v[r];
    }
  }

  if (Ypk) {
    // lane L and L^1 hold adjacent couts of the same pixels -> pack pairs
    unsigned d[8];
    #pragma unroll
    for (int r = 0; r < 8; ++r) {
      const float nb = __shfl_xor(v[r], 1, 32);
      const float lo = (nL & 1) ? nb   : v[r];
      const float hi = (nL & 1) ? v[r] : nb;
      d[r] = pk2bf(lo, hi);
    }
    unsigned* Pp = Ypk + (long)b * ypkB + (long)(cO >> 1) * HW + pixOff;
    if ((nL & 1) == 0) {
      if (aligned) {
        *(uint4*)Pp       = make_uint4(d[0], d[1], d[2], d[3]);
        *(uint4*)(Pp + 4) = make_uint4(d[4], d[5], d[6], d[7]);
      } else {
        #pragma unroll
        for (int r = 0; r < 8; ++r)
          if (pixOff + r < HW) Pp[r] = d[r];
      }
    }
  }
}

// fp32 NCHW -> PK bf16 (zero-fills channel padding)
__global__ void pack_pk_kernel(const float* __restrict__ X, unsigned* __restrict__ P,
                               int Creal, int Cpad, int HW, long xB, long pB)
{
  const long total = (long)(Cpad >> 1) * HW;
  const long i = (long)blockIdx.x * blockDim.x + threadIdx.x;
  const int  b = blockIdx.y;
  if (i >= total) return;
  const int p   = (int)(i / HW);
  const int pix = (int)(i - (long)p * HW);
  const int c0  = 2 * p;
  const float x0 = (c0     < Creal) ? X[(long)b * xB + (long)c0 * HW + pix]       : 0.f;
  const float x1 = (c0 + 1 < Creal) ? X[(long)b * xB + (long)(c0 + 1) * HW + pix] : 0.f;
  P[(long)b * pB + i] = pk2bf(x0, x1);
}

// zero a PK pad region (channel pairs [off, off+cnt) per batch)
__global__ void zero_pk_kernel(unsigned* __restrict__ P, long off, long cnt, long pB)
{
  const long i = (long)blockIdx.x * blockDim.x + threadIdx.x;
  const int  b = blockIdx.y;
  if (i < cnt) P[(long)b * pB + off + i] = 0u;
}

// ROI crop (identity resize) -> fp32 patch into d_out AND PK bf16 patch
__global__ void roi_pack_kernel(const float* __restrict__ src, float* __restrict__ dstF,
                                unsigned* __restrict__ dstPk,
                                const int* __restrict__ locs, int roiIdx, int divv,
                                int C, int Hs, int Ws, int S,
                                long srcB, long dstFB, long pkB)
{
  const int SS    = S * S;
  const int total = (C >> 1) * SS;
  const int i = blockIdx.x * blockDim.x + threadIdx.x;
  const int b = blockIdx.y;
  if (i >= total) return;
  const int x0 = locs[b * 16 + roiIdx * 4 + 0] / divv;
  const int y0 = locs[b * 16 + roiIdx * 4 + 1] / divv;
  const int p   = i / SS;
  const int rem = i - p * SS;
  const int y   = rem / S;
  const int x   = rem - y * S;
  const float* sp = src + (long)b * srcB + ((long)(2 * p) * Hs + (y0 + y)) * Ws + (x0 + x);
  const float v0 = sp[0];
  const float v1 = sp[(long)Hs * Ws];
  dstF[(long)b * dstFB + (long)(2 * p)     * SS + rem] = v0;
  dstF[(long)b * dstFB + (long)(2 * p + 1) * SS + rem] = v1;
  dstPk[(long)b * pkB + (long)p * SS + rem] = pk2bf(v0, v1);
}

// ---------------------------------------------------------------------------
// Host-side orchestration
// ---------------------------------------------------------------------------

// per-batch flat offsets inside one d_out row (concat order from reference)
static constexpr long OFF_F2     = 0;
static constexpr long OFF_F4     = 4194304;
static constexpr long OFF_F6     = 6291456;
static constexpr long OFF_LE256  = 7340032;
static constexpr long OFF_RE256  = 7442432;
static constexpr long OFF_MO256  = 7544832;
static constexpr long OFF_LE128  = 7738432;
static constexpr long OFF_RE128  = 7789632;
static constexpr long OFF_MO128  = 7840832;
static constexpr long OFF_LE64   = 7934144;
static constexpr long OFF_RE64   = 7959744;
static constexpr long OFF_MO64   = 7985344;
static constexpr long OFF_LE256Q = 8028608;
static constexpr long OFF_RE256Q = 8054208;
static constexpr long OFF_MO256Q = 8079808;
static constexpr long OFF_LE128Q = 8128208;
static constexpr long OFF_RE128Q = 8141008;
static constexpr long OFF_MO128Q = 8153808;
static constexpr long OFF_LE64Q  = 8177136;
static constexpr long OFF_RE64Q  = 8183536;
static constexpr long OFF_MO64Q  = 8189936;
static constexpr long ROW        = 8200752;   // total per-batch row length

static inline void launch_conv(hipStream_t s, const unsigned* Xpk, const float* W,
                               const float* Bi, const float* Res,
                               float* Yf, unsigned* Ypk,
                               int CinR, int CinP, int Cout, int Hin, int Win,
                               int stride, int dil, int leaky,
                               long xB, long yfB, long ypkB, long resB)
{
  const int Hout = (Hin - 1) / stride + 1;
  const int Wout = (Win - 1) / stride + 1;
  dim3 grid((Hout * Wout + 63) / 64, Cout / 16, 2);
  #define LC(TW) conv3x3_pk_wmma<TW><<<grid, 128, 0, s>>>(Xpk, W, Bi, Res, Yf, Ypk, \
      CinR, CinP, Cout, Hin, Win, Hout, Wout, stride, dil, leaky, xB, yfB, ypkB, resB)
  if      (Hin == 512 && Win == 512) LC(512);
  else if (Hin == 256 && Win == 256) LC(256);
  else if (Hin == 128 && Win == 128) LC(128);
  else if (Hin == 64  && Win == 64)  LC(64);
  else                               LC(0);
  #undef LC
}

extern "C" void kernel_launch(void* const* d_in, const int* in_sizes, int n_in,
                              void* d_out, int out_size, void* d_ws, size_t ws_size,
                              hipStream_t stream) {
  (void)in_sizes; (void)n_in; (void)out_size; (void)ws_size;

  const float* img  = (const float*)d_in[0];
  const int*   locs = (const int*)d_in[1];
  auto W0 = [&](int blk) { return (const float*)d_in[2 + blk * 4 + 0]; };
  auto b0 = [&](int blk) { return (const float*)d_in[2 + blk * 4 + 1]; };
  auto W1 = [&](int blk) { return (const float*)d_in[2 + blk * 4 + 2]; };
  auto b1 = [&](int blk) { return (const float*)d_in[2 + blk * 4 + 3]; };
  float* out = (float*)d_out;

  // ---- workspace carve-up (4-byte units), bump allocator ------------------
  const long PBimg = 4194304;  // 16 pairs * 512^2
  const long PB1 = 2097152, FB1 = 4194304;   // 32 pairs / 64 ch * 256^2
  const long PB2 = 1048576, FB2 = 2097152;   // 64 pairs / 128 ch * 128^2
  const long PB3 = 524288,  FB3 = 1048576;   // 128 pairs / 256 ch * 64^2
  char* wsb = (char*)d_ws;
  long  cur = 0;
  auto allocU = [&](long n) { unsigned* p = (unsigned*)(wsb + cur * 4); cur += n; return p; };
  auto allocF = [&](long n) { float*    p = (float*)   (wsb + cur * 4); cur += n; return p; };

  unsigned* imgPK = allocU(2 * PBimg);
  unsigned* pkT1 = allocU(2 * PB1); unsigned* pkA1 = allocU(2 * PB1); unsigned* pkB1 = allocU(2 * PB1);
  float*    f1_0 = allocF(2 * FB1); float*    f2_0 = allocF(2 * FB1);
  unsigned* pkT2 = allocU(2 * PB2); unsigned* pkA2 = allocU(2 * PB2); unsigned* pkB2 = allocU(2 * PB2);
  float*    f3_0 = allocF(2 * FB2); float*    f4_0 = allocF(2 * FB2);
  unsigned* pkT3 = allocU(2 * PB3); unsigned* pkA3 = allocU(2 * PB3); unsigned* pkB3 = allocU(2 * PB3);
  float*    f5_0 = allocF(2 * FB3); float*    f6_0 = allocF(2 * FB3);
  unsigned* roiPK  = allocU(262144);
  unsigned* qtmpPK = allocU(131072);

  // ---- pack input image: fp32 NCHW (C=3) -> PK bf16 padded to 32 ch ------
  {
    const long total = 16l * 262144;
    dim3 grid((unsigned)((total + 255) / 256), 2);
    pack_pk_kernel<<<grid, 256, 0, stream>>>(img, imgPK, 3, 32, 262144, 786432, PBimg);
  }

  // -------- stage 1: 64ch @ 256x256 ---------------------------------------
  launch_conv(stream, imgPK, W0(0), b0(0), nullptr, nullptr, pkT1, 3, 32, 64, 512, 512, 2, 1, 1, PBimg, 0, PB1, 0);
  launch_conv(stream, pkT1,  W1(0), b1(0), nullptr, f1_0,    pkA1, 64, 64, 64, 256, 256, 1, 1, 0, PB1, FB1, PB1, 0); // f1_0
  launch_conv(stream, pkA1,  W0(1), b0(1), nullptr, nullptr, pkT1, 64, 64, 64, 256, 256, 1, 5, 1, PB1, 0, PB1, 0);
  launch_conv(stream, pkT1,  W1(1), b1(1), f1_0,    nullptr, pkB1, 64, 64, 64, 256, 256, 1, 3, 0, PB1, 0, PB1, FB1); // f1_1
  launch_conv(stream, pkB1,  W0(2), b0(2), nullptr, nullptr, pkT1, 64, 64, 64, 256, 256, 1, 1, 1, PB1, 0, PB1, 0);
  launch_conv(stream, pkT1,  W1(2), b1(2), nullptr, f2_0,    pkA1, 64, 64, 64, 256, 256, 1, 1, 0, PB1, FB1, PB1, 0); // f2_0
  launch_conv(stream, pkA1,  W0(3), b0(3), nullptr, nullptr, pkT1, 64, 64, 64, 256, 256, 1, 5, 1, PB1, 0, PB1, 0);
  launch_conv(stream, pkT1,  W1(3), b1(3), f2_0, out + OFF_F2, pkB1, 64, 64, 64, 256, 256, 1, 3, 0, PB1, ROW, PB1, FB1); // f2_1

  // -------- stage 2: 128ch @ 128x128 --------------------------------------
  launch_conv(stream, pkB1,  W0(4), b0(4), nullptr, nullptr, pkT2, 64, 64, 128, 256, 256, 2, 1, 1, PB1, 0, PB2, 0);
  launch_conv(stream, pkT2,  W1(4), b1(4), nullptr, f3_0,    pkA2, 128, 128, 128, 128, 128, 1, 1, 0, PB2, FB2, PB2, 0); // f3_0
  launch_conv(stream, pkA2,  W0(5), b0(5), nullptr, nullptr, pkT2, 128, 128, 128, 128, 128, 1, 3, 1, PB2, 0, PB2, 0);
  launch_conv(stream, pkT2,  W1(5), b1(5), f3_0,    nullptr, pkB2, 128, 128, 128, 128, 128, 1, 1, 0, PB2, 0, PB2, FB2); // f3_1
  launch_conv(stream, pkB2,  W0(6), b0(6), nullptr, nullptr, pkT2, 128, 128, 128, 128, 128, 1, 1, 1, PB2, 0, PB2, 0);
  launch_conv(stream, pkT2,  W1(6), b1(6), nullptr, f4_0,    pkA2, 128, 128, 128, 128, 128, 1, 1, 0, PB2, FB2, PB2, 0); // f4_0
  launch_conv(stream, pkA2,  W0(7), b0(7), nullptr, nullptr, pkT2, 128, 128, 128, 128, 128, 1, 3, 1, PB2, 0, PB2, 0);
  launch_conv(stream, pkT2,  W1(7), b1(7), f4_0, out + OFF_F4, pkB2, 128, 128, 128, 128, 128, 1, 1, 0, PB2, ROW, PB2, FB2); // f4_1

  // -------- stage 3: 256ch @ 64x64 ----------------------------------------
  launch_conv(stream, pkB2,  W0(8), b0(8), nullptr, nullptr, pkT3, 128, 128, 256, 128, 128, 2, 1, 1, PB2, 0, PB3, 0);
  launch_conv(stream, pkT3,  W1(8), b1(8), nullptr, f5_0,    pkA3, 256, 256, 256, 64, 64, 1, 1, 0, PB3, FB3, PB3, 0); // f5_0
  launch_conv(stream, pkA3,  W0(9), b0(9), nullptr, nullptr, pkT3, 256, 256, 256, 64, 64, 1, 1, 1, PB3, 0, PB3, 0);
  launch_conv(stream, pkT3,  W1(9), b1(9), f5_0,    nullptr, pkB3, 256, 256, 256, 64, 64, 1, 1, 0, PB3, 0, PB3, FB3); // f5_1
  launch_conv(stream, pkB3,  W0(10), b0(10), nullptr, nullptr, pkT3, 256, 256, 256, 64, 64, 1, 1, 1, PB3, 0, PB3, 0);
  launch_conv(stream, pkT3,  W1(10), b1(10), nullptr, f6_0,    pkA3, 256, 256, 256, 64, 64, 1, 1, 0, PB3, FB3, PB3, 0); // f6_0
  launch_conv(stream, pkA3,  W0(11), b0(11), nullptr, nullptr, pkT3, 256, 256, 256, 64, 64, 1, 1, 1, PB3, 0, PB3, 0);
  launch_conv(stream, pkT3,  W1(11), b1(11), f6_0, out + OFF_F6, nullptr, 256, 256, 256, 64, 64, 1, 1, 0, PB3, ROW, 0, FB3); // f6_1

  // -------- ROI + Q blocks -------------------------------------------------
  struct QD { int blk; long srcOff; int Hs; int C; int divv; int roi;
              long roiOff; long qOff; int S; int Cq; int CqPad; };
  const QD qd[9] = {
    {12, OFF_F2, 256, 64,  2, 0, OFF_LE256, OFF_LE256Q, 40, 16, 32},
    {13, OFF_F2, 256, 64,  2, 1, OFF_RE256, OFF_RE256Q, 40, 16, 32},
    {14, OFF_F2, 256, 64,  2, 3, OFF_MO256, OFF_MO256Q, 55, 16, 32},
    {15, OFF_F4, 128, 128, 4, 0, OFF_LE128, OFF_LE128Q, 20, 32, 32},
    {16, OFF_F4, 128, 128, 4, 1, OFF_RE128, OFF_RE128Q, 20, 32, 32},
    {17, OFF_F4, 128, 128, 4, 3, OFF_MO128, OFF_MO128Q, 27, 32, 32},
    {18, OFF_F6, 64,  256, 8, 0, OFF_LE64,  OFF_LE64Q,  10, 64, 64},
    {19, OFF_F6, 64,  256, 8, 1, OFF_RE64,  OFF_RE64Q,  10, 64, 64},
    {20, OFF_F6, 64,  256, 8, 3, OFF_MO64,  OFF_MO64Q,  13, 64, 64},
  };
  for (int i = 0; i < 9; ++i) {
    const QD& q = qd[i];
    const int  SS   = q.S * q.S;
    const long rpkB = (long)(q.C >> 1) * SS;       // ROI PK per-batch stride
    const long qpB  = (long)(q.CqPad >> 1) * SS;   // q-intermediate PK stride

    {  // ROI slice: fp32 patch into d_out + PK patch for the Q conv
      const int total = (q.C >> 1) * SS;
      dim3 grid((total + 255) / 256, 2);
      roi_pack_kernel<<<grid, 256, 0, stream>>>(out + q.srcOff, out + q.roiOff, roiPK,
                                                locs, q.roi, q.divv, q.C, q.Hs, q.Hs,
                                                q.S, ROW, ROW, rpkB);
    }
    if (q.CqPad > q.Cq) {  // zero the PK channel padding read by q-conv1
      const long off = (long)(q.Cq >> 1) * SS;
      const long cnt = (long)((q.CqPad - q.Cq) >> 1) * SS;
      dim3 grid((unsigned)((cnt + 255) / 256), 2);
      zero_pk_kernel<<<grid, 256, 0, stream>>>(qtmpPK, off, cnt, qpB);
    }
    launch_conv(stream, roiPK, W0(q.blk), b0(q.blk), nullptr, nullptr, qtmpPK,
                q.C, q.C, q.Cq, q.S, q.S, 1, 1, 1, rpkB, 0, qpB, 0);
    launch_conv(stream, qtmpPK, W1(q.blk), b1(q.blk), nullptr, out + q.qOff, nullptr,
                q.Cq, q.CqPad, q.Cq, q.S, q.S, 1, 1, 0, qpB, ROW, 0, 0);
  }
}
